// RegimeAwareStudent_62989990363249
// MI455X (gfx1250) — matrix operand order repaired and checked by reading, not compiled, not run on hardware
//
#include <hip/hip_runtime.h>

// ---------------------------------------------------------------------------
// RegimeAwareStudent fused kernel for MI455X (gfx1250), bf16 WMMA path.
//   B=524288, D=128, H=256, H2=128, H4=64, R=3
// v2: M=32 per wave (two 16-row subtiles share each B fragment) to halve
//     L2 weight traffic and double the WMMA:VMEM ratio.
// ---------------------------------------------------------------------------

typedef __attribute__((ext_vector_type(16))) __bf16 bf16x16;
typedef __attribute__((ext_vector_type(8)))  float  f32x8;

union Frag16 { bf16x16 v; uint4 u[2]; };

// B fragment: two 8-bf16 chunks at +0 and +16 elems (lane's column, from [N][K] layout)
__device__ __forceinline__ bf16x16 load_b_frag(const __bf16* p) {
  Frag16 f;
  const uint4* q = (const uint4*)p;
  f.u[0] = q[0];
  f.u[1] = q[2];
  return f.v;
}

// A fragment from LDS row-major bf16: two 16-byte chunks at +0 and +16 elems
__device__ __forceinline__ bf16x16 load_a_frag_lds(const __bf16* p) {
  Frag16 f;
  f.u[0] = *(const uint4*)(p);
  f.u[1] = *(const uint4*)(p + 16);
  return f.v;
}

// A fragment built from f32 global memory (x), rounding to bf16:
// elems 0..7 = K kb..kb+7, elems 8..15 = K kb+16..kb+23
__device__ __forceinline__ bf16x16 make_a_from_f32(const float* p) {
  bf16x16 a;
#pragma unroll
  for (int e = 0; e < 8; ++e) a[e] = (__bf16)p[e];
#pragma unroll
  for (int e = 0; e < 8; ++e) a[8 + e] = (__bf16)p[16 + e];
  return a;
}

// ---------------------------------------------------------------------------
// Prep: bf16-convert + transpose weights into workspace:
//   ws[0      .. 32767 ]  W1t[n][k]  n<256,k<128   (from fe_w1[k][n], [128,256])
//   ws[32768  .. 65535 ]  W2t[n][k]  n<128,k<256   (from fe_w2[k][n], [256,128])
//   ws[65536  .. 212991]  E1t[r][n][k] n<256,k<192 (from ew1[r][k][n], [3,192,256])
// ---------------------------------------------------------------------------
__global__ __launch_bounds__(256) void regime_prep_kernel(
    const float* __restrict__ fw1, const float* __restrict__ fw2,
    const float* __restrict__ ew1, __bf16* __restrict__ ws) {
  int t = blockIdx.x * 256 + threadIdx.x;
  if (t < 32768) {
    int n = t >> 7, k = t & 127;
    ws[t] = (__bf16)fw1[k * 256 + n];
    return;
  }
  t -= 32768;
  if (t < 32768) {
    int n = t >> 8, k = t & 255;
    ws[32768 + n * 256 + k] = (__bf16)fw2[k * 128 + n];
    return;
  }
  t -= 32768;
  if (t < 147456) {
    int r = t / 49152, rem = t % 49152;
    int n = rem / 192, k = rem % 192;
    ws[65536 + t] = (__bf16)ew1[(r * 192 + k) * 256 + n];
  }
}

// ---------------------------------------------------------------------------
// Fused: GEMM1 -> ReLU -> GEMM2 -> ReLU -> concat(emb gather) -> 3 experts
// 4 waves/block, 32 rows/wave (2 subtiles), 128 rows/block, 4096 blocks.
// LDS: 4*(32*256 + 32*192)*2 = 112 KB.
// ---------------------------------------------------------------------------
__global__ __launch_bounds__(128, 1) void regime_fused_kernel(
    const float* __restrict__ x,    const int* __restrict__ rid,
    const float* __restrict__ fb1,  const float* __restrict__ fb2,
    const float* __restrict__ emb,  const float* __restrict__ eb1,
    const float* __restrict__ ew2,  const float* __restrict__ eb2,
    const __bf16* __restrict__ W1t, const __bf16* __restrict__ W2t,
    const __bf16* __restrict__ E1t, float* __restrict__ out) {
  __shared__ __bf16 sh[4][32 * 256];  // per-wave h staging (bf16 row-major)
  __shared__ __bf16 sc[4][32 * 192];  // per-wave combined staging

  const int  wave = threadIdx.x >> 5;
  const int  lane = threadIdx.x & 31;
  const int  hi   = lane >> 4;   // lane half selects K sub-range of fragments
  const int  l    = lane & 15;   // row (A) / column (B,C) index within tile
  const long row0 = (long)blockIdx.x * 128 + wave * 32;

  __bf16* hbuf = sh[wave];
  __bf16* cbuf = sc[wave];

  // ---------------- Stage 1: h = relu(x @ fe_w1 + fe_b1), K=128, N=256 -----
  bf16x16 a1[2][4];
#pragma unroll
  for (int s = 0; s < 2; ++s) {
    const float* xr = x + (row0 + s * 16 + l) * 128;
#pragma unroll
    for (int kc = 0; kc < 4; ++kc)
      a1[s][kc] = make_a_from_f32(xr + kc * 32 + hi * 8);
  }
  for (int nc = 0; nc < 16; ++nc) {
    f32x8 c0 = {}, c1 = {};
#pragma unroll
    for (int kc = 0; kc < 4; ++kc) {
      bf16x16 b = load_b_frag(W1t + (nc * 16 + l) * 128 + kc * 32 + hi * 8);
      c0 = __builtin_amdgcn_wmma_f32_16x16x32_bf16(false, a1[0][kc], false, b,
                                                   (short)0, c0, false, false);
      c1 = __builtin_amdgcn_wmma_f32_16x16x32_bf16(false, a1[1][kc], false, b,
                                                   (short)0, c1, false, false);
    }
    const float bias = fb1[nc * 16 + l];
#pragma unroll
    for (int i = 0; i < 8; ++i) {
      float v0 = c0[i] + bias; v0 = v0 > 0.f ? v0 : 0.f;
      float v1 = c1[i] + bias; v1 = v1 > 0.f ? v1 : 0.f;
      hbuf[(hi * 8 + i) * 256 + nc * 16 + l]        = (__bf16)v0;
      hbuf[(16 + hi * 8 + i) * 256 + nc * 16 + l]   = (__bf16)v1;
    }
  }
  __syncthreads();

  // ---------------- Stage 2: features = relu(h @ fe_w2 + fe_b2), K=256, N=128
  for (int nc = 0; nc < 8; ++nc) {
    f32x8 c0 = {}, c1 = {};
#pragma unroll
    for (int kc = 0; kc < 8; ++kc) {
      bf16x16 b  = load_b_frag(W2t + (nc * 16 + l) * 256 + kc * 32 + hi * 8);
      bf16x16 a0 = load_a_frag_lds(hbuf + l * 256 + kc * 32 + hi * 8);
      bf16x16 a1f = load_a_frag_lds(hbuf + (16 + l) * 256 + kc * 32 + hi * 8);
      c0 = __builtin_amdgcn_wmma_f32_16x16x32_bf16(false, a0, false, b,
                                                   (short)0, c0, false, false);
      c1 = __builtin_amdgcn_wmma_f32_16x16x32_bf16(false, a1f, false, b,
                                                   (short)0, c1, false, false);
    }
    const float bias = fb2[nc * 16 + l];
#pragma unroll
    for (int i = 0; i < 8; ++i) {
      float v0 = c0[i] + bias; v0 = v0 > 0.f ? v0 : 0.f;
      float v1 = c1[i] + bias; v1 = v1 > 0.f ? v1 : 0.f;
      cbuf[(hi * 8 + i) * 192 + nc * 16 + l]      = (__bf16)v0;
      cbuf[(16 + hi * 8 + i) * 192 + nc * 16 + l] = (__bf16)v1;
    }
  }
  // regime embedding gather -> combined cols 128..191 (each lane fills 2 cols)
  for (int r = 0; r < 32; ++r) {
    const int    id = rid[row0 + r];
    const float* e  = emb + id * 64 + lane * 2;
    cbuf[r * 192 + 128 + lane * 2 + 0] = (__bf16)e[0];
    cbuf[r * 192 + 128 + lane * 2 + 1] = (__bf16)e[1];
  }
  __syncthreads();

  // ---------------- Stage 3: experts, K=192, N=256, then dot with ew2 ------
  for (int rr = 0; rr < 3; ++rr) {
    float acc0[8], acc1[8];
#pragma unroll
    for (int i = 0; i < 8; ++i) { acc0[i] = 0.f; acc1[i] = 0.f; }
    const __bf16* Er = E1t + rr * (256 * 192);
    for (int nc = 0; nc < 16; ++nc) {
      f32x8 c0 = {}, c1 = {};
#pragma unroll
      for (int kc = 0; kc < 6; ++kc) {
        bf16x16 b  = load_b_frag(Er + (nc * 16 + l) * 192 + kc * 32 + hi * 8);
        bf16x16 a0 = load_a_frag_lds(cbuf + l * 192 + kc * 32 + hi * 8);
        bf16x16 a1f = load_a_frag_lds(cbuf + (16 + l) * 192 + kc * 32 + hi * 8);
        c0 = __builtin_amdgcn_wmma_f32_16x16x32_bf16(false, a0, false, b,
                                                     (short)0, c0, false, false);
        c1 = __builtin_amdgcn_wmma_f32_16x16x32_bf16(false, a1f, false, b,
                                                     (short)0, c1, false, false);
      }
      // bias + ReLU + fold the N=1 projection into a per-lane partial dot
      const float b1v = eb1[rr * 256 + nc * 16 + l];
      const float w2v = ew2[rr * 256 + nc * 16 + l];
#pragma unroll
      for (int i = 0; i < 8; ++i) {
        float v0 = c0[i] + b1v; v0 = v0 > 0.f ? v0 : 0.f;
        float v1 = c1[i] + b1v; v1 = v1 > 0.f ? v1 : 0.f;
        acc0[i] = fmaf(v0, w2v, acc0[i]);
        acc1[i] = fmaf(v1, w2v, acc1[i]);
      }
    }
    // reduce over the 16 lanes of each half (rows 0..7 in lanes 0-15, 8..15 in 16-31)
#pragma unroll
    for (int i = 0; i < 8; ++i) {
      acc0[i] += __shfl_xor(acc0[i], 1, 32);
      acc0[i] += __shfl_xor(acc0[i], 2, 32);
      acc0[i] += __shfl_xor(acc0[i], 4, 32);
      acc0[i] += __shfl_xor(acc0[i], 8, 32);
      acc1[i] += __shfl_xor(acc1[i], 1, 32);
      acc1[i] += __shfl_xor(acc1[i], 2, 32);
      acc1[i] += __shfl_xor(acc1[i], 4, 32);
      acc1[i] += __shfl_xor(acc1[i], 8, 32);
    }
    if (l == 0) {  // lanes 0 and 16 own rows [0..7] and [8..15] of each subtile
      const float b2v = eb2[rr];
#pragma unroll
      for (int i = 0; i < 8; ++i) {
        const long g0 = row0 + hi * 8 + i;
        const long g1 = row0 + 16 + hi * 8 + i;
        if (rid[g0] == rr) out[g0] = acc0[i] + b2v;
        if (rid[g1] == rr) out[g1] = acc1[i] + b2v;
      }
    }
  }
}

// ---------------------------------------------------------------------------
extern "C" void kernel_launch(void* const* d_in, const int* in_sizes, int n_in,
                              void* d_out, int out_size, void* d_ws, size_t ws_size,
                              hipStream_t stream) {
  (void)in_sizes; (void)n_in; (void)out_size; (void)ws_size;
  const float* x   = (const float*)d_in[0];
  const int*   rid = (const int*)d_in[1];
  const float* fw1 = (const float*)d_in[2];
  const float* fb1 = (const float*)d_in[3];
  const float* fw2 = (const float*)d_in[4];
  const float* fb2 = (const float*)d_in[5];
  const float* emb = (const float*)d_in[6];
  const float* ew1 = (const float*)d_in[7];
  const float* eb1 = (const float*)d_in[8];
  const float* ew2 = (const float*)d_in[9];
  const float* eb2 = (const float*)d_in[10];
  float*  out = (float*)d_out;
  __bf16* ws  = (__bf16*)d_ws;  // 212992 bf16 = 416 KB

  // weight convert+transpose (212992 elements, 832 * 256 covers exactly)
  regime_prep_kernel<<<832, 256, 0, stream>>>(fw1, fw2, ew1, ws);

  // fused pipeline: 128 rows per block, 524288 / 128 = 4096 blocks
  regime_fused_kernel<<<4096, 128, 0, stream>>>(
      x, rid, fb1, fb2, emb, eb1, ew2, eb2,
      ws, ws + 32768, ws + 65536, out);
}
